// UnpoolSubDivision_15960098472030
// MI455X (gfx1250) — matrix operand order repaired and checked by reading, not compiled, not run on hardware
//
#include <hip/hip_runtime.h>
#include <stdint.h>

#define B_ 8
#define N_ 8192
#define D_ 128
#define K_ 4
#define TILES_ (N_ / 16)

typedef __attribute__((ext_vector_type(16))) __bf16 v16bf;
typedef __attribute__((ext_vector_type(8)))  __bf16 v8bf;
typedef __attribute__((ext_vector_type(8)))  float  v8f;
typedef __attribute__((ext_vector_type(4)))  float  f4;
typedef __attribute__((ext_vector_type(4)))  unsigned int v4u;
typedef __attribute__((ext_vector_type(8)))  int v8i_;
typedef __attribute__((ext_vector_type(4)))  int v4i_;

union V16U { v16bf v; v8bf h[2]; };

#define KEY_BIG 0x7F000000u   // ~1.7e38, finite sentinel (decodes to idx 0)
#define IDX_MASK 0x1FFFu      // 13 bits, N = 8192

#if __has_builtin(__builtin_amdgcn_tensor_load_to_lds)
#define HAVE_TDM 1
#else
#define HAVE_TDM 0
#endif

__device__ __forceinline__ float med3f(float a, float b, float c) {
#if __has_builtin(__builtin_amdgcn_fmed3f)
    return __builtin_amdgcn_fmed3f(a, b, c);
#else
    return fmaxf(fminf(a, b), fminf(fmaxf(a, b), c));
#endif
}

// Branchless insert of key kv into ascending sorted quad k[0..3] (keeps 4 smallest).
__device__ __forceinline__ void upd4(float kv, float k[4]) {
    float o0 = k[0], o1 = k[1], o2 = k[2], o3 = k[3];
    k[0] = fminf(o0, kv);
    k[1] = med3f(kv, o0, o1);
    k[2] = med3f(kv, o1, o2);
    k[3] = med3f(kv, o2, o3);
}

// Pack candidate: distance key with index in the low 13 mantissa bits.
__device__ __forceinline__ float packKey(float val, unsigned col) {
    return __uint_as_float((__float_as_uint(val) & ~IDX_MASK) | col);
}

__device__ __forceinline__ void wait_tensorcnt2() {
#if __has_builtin(__builtin_amdgcn_s_wait_tensorcnt)
    __builtin_amdgcn_s_wait_tensorcnt(2);
#else
    asm volatile("s_wait_tensorcnt 0x2" ::: "memory");
#endif
}

#if HAVE_TDM
// TDM: DMA one 16-row x 256B tile (row stride 256B) from global into LDS.
// D# per CDNA5 ISA ch.8: 2D tensor, data_size=8B units, groups 2/3 zero (<=2D).
// This toolchain exposes the 6-arg builtin form:
//   (uint32x4 g0, int32x8 g1, int32x4 g2, int32x4 g3, int32x8, i32 cpol)
__device__ __forceinline__ void tdm_load_tile(const unsigned short* gsrc, unsigned ldsByteAddr) {
    unsigned long long ga = (unsigned long long)(uintptr_t)gsrc;
    v4u g0;
    g0.x = 1u;                                                   // count=1, user D#
    g0.y = ldsByteAddr;                                          // lds_addr
    g0.z = (unsigned)(ga & 0xFFFFFFFFu);                         // global_addr[31:0]
    g0.w = ((unsigned)(ga >> 32) & 0x01FFFFFFu) | 0x80000000u;   // addr[56:32] | type=2
    v8i_ g1;
    g1[0] = 0x00030000;            // wg_mask=0, data_size=3 (8B), no flags
    g1[1] = (int)(32u << 16);      // tensor_dim0 = 32 units (256B row), lo16 @ [63:48]
    g1[2] = (int)(16u << 16);      // tensor_dim0 hi | tensor_dim1 = 16 rows (lo16)
    g1[3] = (int)(32u << 16);      // tensor_dim1 hi | tile_dim0 = 32 units
    g1[4] = 16;                    // tile_dim1 = 16 rows, tile_dim2 = 0
    g1[5] = 32;                    // tensor_dim0_stride = 32 units (256B)
    g1[6] = 0;                     // stride0 hi | tensor_dim1_stride lo
    g1[7] = 0;                     // tensor_dim1_stride hi
    v4i_ z4 = {0, 0, 0, 0};
    v8i_ z8 = {0, 0, 0, 0, 0, 0, 0, 0};
    __builtin_amdgcn_tensor_load_to_lds(g0, g1, z4, z4, z8, 0);
}
#endif

// Split f32 -> bf16 hi/lo planes (RNE) and compute per-point squared norms.
__global__ void knn_prep(const float* __restrict__ x,
                         unsigned short* __restrict__ xh,
                         unsigned short* __restrict__ xl,
                         float* __restrict__ sq) {
    int p = blockIdx.x;            // point index in [0, B*N)
    int d = threadIdx.x;           // 0..127
    size_t off = (size_t)p * D_ + d;
    float f = x[off];
    unsigned int u  = __float_as_uint(f);
    unsigned int rh = u + 0x7FFFu + ((u >> 16) & 1u);
    unsigned short hb = (unsigned short)(rh >> 16);
    float hf = __uint_as_float((unsigned int)hb << 16);
    float lf = f - hf;
    unsigned int ul = __float_as_uint(lf);
    unsigned int rl = ul + 0x7FFFu + ((ul >> 16) & 1u);
    unsigned short lb = (unsigned short)(rl >> 16);
    xh[off] = hb;
    xl[off] = lb;

    __shared__ float red[128];
    red[d] = f * f;
    __syncthreads();
    #pragma unroll
    for (int s = 64; s > 0; s >>= 1) {
        if (d < s) red[d] += red[d + s];
        __syncthreads();
    }
    if (d == 0) sq[p] = red[0];
}

// 4-wave workgroup, 128 query rows (32 per wave). B tiles staged to LDS by the
// Tensor Data Mover (double-buffered, overlapped with WMMA), bf16x3 Gram via
// v_wmma_f32_16x16x32_bf16, branchless med3 top-4 with packed keys, then
// gather + midpoint output.
__global__ void __launch_bounds__(128)
knn_main(const float* __restrict__ x,
         const unsigned short* __restrict__ xh,
         const unsigned short* __restrict__ xl,
         const float* __restrict__ sq,
         float* __restrict__ out) {
    const int tid  = threadIdx.x;
    const int wid  = tid >> 5;
    const int lane = tid & 31;
    const int b    = blockIdx.y;
    const int rowBase = blockIdx.x * 128;
    const int hl   = lane >> 4;
    const int l16  = lane & 15;

    const size_t bofs = (size_t)b * N_ * D_;
    const unsigned short* xhB = xh + bofs;
    const unsigned short* xlB = xl + bofs;

    __shared__ __attribute__((aligned(16))) unsigned short sB[2][2][16 * D_]; // [parity][hi/lo][row*128+k]
    __shared__ float s_k[128][16][4];
    __shared__ int   s_fi[128][4];

    // ---- Preload A fragments for this wave's two 16-row sets.
    v16bf aH[2][4], aL[2][4];
    #pragma unroll
    for (int s2 = 0; s2 < 2; ++s2) {
        const unsigned short* ah = xhB + (size_t)(rowBase + wid * 32 + s2 * 16 + l16) * D_;
        const unsigned short* al = xlB + (size_t)(rowBase + wid * 32 + s2 * 16 + l16) * D_;
        #pragma unroll
        for (int k = 0; k < 4; ++k) {
            int ab = k * 32 + hl * 16;
            V16U u0, u1;
            u0.h[0] = *reinterpret_cast<const v8bf*>(ah + ab);
            u0.h[1] = *reinterpret_cast<const v8bf*>(ah + ab + 8);
            aH[s2][k] = u0.v;
            u1.h[0] = *reinterpret_cast<const v8bf*>(al + ab);
            u1.h[1] = *reinterpret_cast<const v8bf*>(al + ab + 8);
            aL[s2][k] = u1.v;
        }
    }

    // Per-lane sorted top-4 keys for the 16 rows this lane owns (8 per set).
    float bk[16][4];
    #pragma unroll
    for (int r = 0; r < 16; ++r)
        #pragma unroll
        for (int j = 0; j < 4; ++j)
            bk[r][j] = __uint_as_float(KEY_BIG);

    const float* sqb = sq + (size_t)b * N_;

#if HAVE_TDM
    if (wid == 0) {   // prologue: DMA tile 0 into buffer 0
        tdm_load_tile(xhB, (unsigned)(uintptr_t)&sB[0][0][0]);
        tdm_load_tile(xlB, (unsigned)(uintptr_t)&sB[0][1][0]);
    }
#endif

    #pragma unroll 1
    for (int t = 0; t < TILES_; ++t) {
        const int colBase = t * 16;
        const unsigned col = (unsigned)(colBase + l16);

#if HAVE_TDM
        {   // DMA next tile into the other buffer (overlaps this tile's math),
            // then wait for this tile's two TDM ops (in-order) and publish.
            const int nb = (t + 1) & (TILES_ - 1);   // wrap: last issue is a dummy
            const int pn = (t + 1) & 1;
            if (wid == 0) {
                tdm_load_tile(xhB + (size_t)(nb * 16) * D_, (unsigned)(uintptr_t)&sB[pn][0][0]);
                tdm_load_tile(xlB + (size_t)(nb * 16) * D_, (unsigned)(uintptr_t)&sB[pn][1][0]);
                wait_tensorcnt2();
            }
        }
        __syncthreads();
        const int p = t & 1;
#else
        {   // fallback: cooperative synchronous staging (512 x 16B chunks / 128 threads)
            #pragma unroll
            for (int c = 0; c < 4; ++c) {
                int ci = tid * 4 + c;
                int plane = ci >> 8;
                int r  = (ci >> 4) & 15;
                int pc = ci & 15;
                const unsigned short* src = (plane ? xlB : xhB) + (size_t)(colBase + r) * D_ + pc * 8;
                *reinterpret_cast<f4*>(&sB[0][plane][r * D_ + pc * 8]) =
                    *reinterpret_cast<const f4*>(src);
            }
        }
        __syncthreads();
        const int p = 0;
#endif

        float sv = sqb[col];
        v8f acc0 = {};
        v8f acc1 = {};
        #pragma unroll
        for (int k = 0; k < 4; ++k) {
            // B fragment (32x16 bf16) straight from LDS: lane = column,
            // contiguous 16-K chunk per half-wave -> 2x ds_load_b128 per plane.
            int bb = k * 32 + hl * 16;
            V16U uh, ulv;
            uh.h[0]  = *reinterpret_cast<const v8bf*>(&sB[p][0][l16 * D_ + bb]);
            uh.h[1]  = *reinterpret_cast<const v8bf*>(&sB[p][0][l16 * D_ + bb + 8]);
            ulv.h[0] = *reinterpret_cast<const v8bf*>(&sB[p][1][l16 * D_ + bb]);
            ulv.h[1] = *reinterpret_cast<const v8bf*>(&sB[p][1][l16 * D_ + bb + 8]);
            // bf16x3: hi*hi + hi*lo + lo*hi, f32 accumulate; both row sets reuse B.
            acc0 = __builtin_amdgcn_wmma_f32_16x16x32_bf16(false, aH[0][k], false, uh.v,  (short)0, acc0, false, false);
            acc0 = __builtin_amdgcn_wmma_f32_16x16x32_bf16(false, aH[0][k], false, ulv.v, (short)0, acc0, false, false);
            acc0 = __builtin_amdgcn_wmma_f32_16x16x32_bf16(false, aL[0][k], false, uh.v,  (short)0, acc0, false, false);
            acc1 = __builtin_amdgcn_wmma_f32_16x16x32_bf16(false, aH[1][k], false, uh.v,  (short)0, acc1, false, false);
            acc1 = __builtin_amdgcn_wmma_f32_16x16x32_bf16(false, aH[1][k], false, ulv.v, (short)0, acc1, false, false);
            acc1 = __builtin_amdgcn_wmma_f32_16x16x32_bf16(false, aL[1][k], false, uh.v,  (short)0, acc1, false, false);
        }
        // Ranking key: sq[col] - 2*dot (row-constant sq[row] dropped; ordering unchanged).
        #pragma unroll
        for (int v = 0; v < 8; ++v) {
            float kv0 = packKey(fmaf(-2.0f, acc0[v], sv), col);
            upd4(kv0, bk[v]);
            float kv1 = packKey(fmaf(-2.0f, acc1[v], sv), col);
            upd4(kv1, bk[8 + v]);
        }
        __syncthreads();   // all waves done with buffer p before its next overwrite
    }

    // ---- Merge 16 per-lane lists per row via LDS (packed keys carry indices).
    #pragma unroll
    for (int v = 0; v < 8; ++v) {
        int row0 = wid * 32 + hl * 8 + v;
        int row1 = wid * 32 + 16 + hl * 8 + v;
        #pragma unroll
        for (int j = 0; j < 4; ++j) {
            s_k[row0][l16][j] = bk[v][j];
            s_k[row1][l16][j] = bk[8 + v][j];
        }
    }
    __syncthreads();

    {   // each of the 128 threads merges one row
        float m[4];
        #pragma unroll
        for (int j = 0; j < 4; ++j) m[j] = __uint_as_float(KEY_BIG);
        for (int c = 0; c < 16; ++c)
            #pragma unroll
            for (int j = 0; j < 4; ++j)
                upd4(s_k[tid][c][j], m);
        #pragma unroll
        for (int j = 0; j < 4; ++j)
            s_fi[tid][j] = (int)(__float_as_uint(m[j]) & IDX_MASK);
    }
    __syncthreads();

    // ---- Gather + midpoint output: out[b, n*4+j, :] = j==0 ? x[i0] : 0.5*(x[i0]+x[ij]).
    const float* xb   = x   + bofs;
    float*       outb = out + (size_t)b * N_ * K_ * D_;
    #pragma unroll 1
    for (int r = 0; r < 32; ++r) {
        int rl = wid * 32 + r;
        int n  = rowBase + rl;
        int i0 = s_fi[rl][0];
        f4 a0 = *reinterpret_cast<const f4*>(xb + (size_t)i0 * D_ + lane * 4);
        *reinterpret_cast<f4*>(outb + ((size_t)n * K_ + 0) * D_ + lane * 4) = a0;
        #pragma unroll
        for (int j = 1; j < 4; ++j) {
            int ij = s_fi[rl][j];
            f4 aj = *reinterpret_cast<const f4*>(xb + (size_t)ij * D_ + lane * 4);
            f4 m  = (a0 + aj) * 0.5f;
            *reinterpret_cast<f4*>(outb + ((size_t)n * K_ + j) * D_ + lane * 4) = m;
        }
    }
}

extern "C" void kernel_launch(void* const* d_in, const int* in_sizes, int n_in,
                              void* d_out, int out_size, void* d_ws, size_t ws_size,
                              hipStream_t stream) {
    (void)in_sizes; (void)n_in; (void)out_size; (void)ws_size;
    const float* x   = (const float*)d_in[0];
    float*       out = (float*)d_out;

    char* ws = (char*)d_ws;
    unsigned short* xh = (unsigned short*)ws;                                   // 16 MB
    unsigned short* xl = (unsigned short*)(ws + (size_t)B_ * N_ * D_ * 2);      // 16 MB
    float*          sq = (float*)        (ws + (size_t)B_ * N_ * D_ * 4);       // 256 KB

    knn_prep<<<B_ * N_, 128, 0, stream>>>(x, xh, xl, sq);
    knn_main<<<dim3(N_ / 128, B_), dim3(128), 0, stream>>>(x, xh, xl, sq, out);
}
